// CrossAttention_13537736917149
// MI455X (gfx1250) — matrix operand order repaired and checked by reading, not compiled, not run on hardware
//
#include <hip/hip_runtime.h>

// CDNA5 (gfx1250) wave32 WMMA cross-attention, vectorized fragment I/O,
// 16x64 output tiles per wave (4 accumulators), double-buffered LDS staging.
// All GEMMs: v_wmma_f32_16x16x32_bf16 (bf16 A/B, f32 accumulate).
//
// Fragment layouts used here (contiguous k order, not the ISA table order), wave32, lane l, m=n=l&15, half=l>>4:
//   A 16x32: row m, slot s holds K = half*16 + s        -> 2x 16B loads
//   B 32x16: col n, slot s holds K = half*16 + s        -> one 32B chunk
//            (pre-swizzled "column panel" storage, see bswz_off)
//   C/D f32: lane l, vgpr r -> M = half*8 + r, N = l&15

typedef __attribute__((ext_vector_type(16))) __bf16 v16bf;
typedef __attribute__((ext_vector_type(8)))  __bf16 v8bf;
typedef __attribute__((ext_vector_type(4)))  __bf16 v4bf;
typedef __attribute__((ext_vector_type(8)))  float  v8f;
typedef __attribute__((ext_vector_type(4)))  float  v4f;

#define N_PTS   65536
#define CH      256
#define BATCH   8
#define KCTX    256
#define CTXD    768
#define HEADS   8
#define DHEAD   64
#define HD      512
#define NPB     8192

__device__ __forceinline__ size_t bswz_off(int k, int n, int ntot) {
    return (size_t)((k >> 5) * (ntot >> 4) + (n >> 4)) * 512
         + (size_t)((((k >> 4) & 1) * 16) + (n & 15)) * 16
         + (k & 15);
}
__device__ __forceinline__ v16bf load_a_frag(const __bf16* rowk0, int half) {
    union { v16bf v; v8bf h[2]; } u;
    u.h[0] = *(const v8bf*)(rowk0 + half * 16);
    u.h[1] = *(const v8bf*)(rowk0 + half * 16 + 8);
    return u.v;
}
__device__ __forceinline__ v16bf load_b_frag(const __bf16* blk, int lane) {
    return *(const v16bf*)(blk + lane * 16);
}

// ---------------------------------------------------------------- cast + swizzle f32 -> bf16 B-panels
__global__ void cast_swz_b(const float* __restrict__ src, __bf16* __restrict__ dst,
                           int ktot, int ntot) {
    int i = blockIdx.x * blockDim.x + threadIdx.x;
    if (i >= ktot * ntot) return;
    int k = i / ntot, n = i % ntot;
    dst[bswz_off(k, n, ntot)] = (__bf16)src[i];
}

// ---------------------------------------------------------------- K/V projection
// [2048,768]f32 @ [768,512]bf16 -> K/V pre-swizzled per (b,h).
// Block: 128 threads (4 waves); wave computes 16x64; A tile double-buffered in LDS.
__global__ void kv_proj(const float* __restrict__ ctx,
                        const __bf16* __restrict__ Wkz, const __bf16* __restrict__ Wvz,
                        __bf16* __restrict__ Kz, __bf16* __restrict__ Vz) {
    __shared__ __bf16 sA[2][16][40];
    const int tid = threadIdx.x;
    const int lane = tid & 31, wv = tid >> 5;
    const int m = lane & 15, half = lane >> 4;
    const int mt = blockIdx.y;                           // 0..127
    const int ntb = blockIdx.x * 16 + wv * 4;            // 16-col tile base, step 4
    const __bf16* Wz = blockIdx.z ? Wvz : Wkz;

    const int srow = tid >> 3, scol = (tid & 7) * 4;
    const float* grow = ctx + (size_t)(mt * 16 + srow) * CTXD + scol;

    v8f acc[4] = {};
    // stage k-chunk 0
    {
        v4f x = *(const v4f*)(grow);
        v4bf y; y.x = (__bf16)x.x; y.y = (__bf16)x.y; y.z = (__bf16)x.z; y.w = (__bf16)x.w;
        *(v4bf*)&sA[0][srow][scol] = y;
    }
    const int NK = CTXD / 32;
    for (int kc = 0; kc < NK; ++kc) {
        __syncthreads();                                  // sA[kc&1] ready; sA[(kc+1)&1] free
        if (kc + 1 < NK) {
            v4f x = *(const v4f*)(grow + (kc + 1) * 32);
            v4bf y; y.x = (__bf16)x.x; y.y = (__bf16)x.y; y.z = (__bf16)x.z; y.w = (__bf16)x.w;
            *(v4bf*)&sA[(kc + 1) & 1][srow][scol] = y;
        }
        v16bf a = load_a_frag(&sA[kc & 1][m][0], half);
        for (int i = 0; i < 4; ++i) {
            v16bf b = load_b_frag(Wz + (size_t)(kc * 32 + ntb + i) * 512, lane);
            acc[i] = __builtin_amdgcn_wmma_f32_16x16x32_bf16(false, a, false, b, (short)0, acc[i], false, false);
        }
    }
    // epilogue: scatter into per-(b,h) swizzled K / V panels
    for (int i = 0; i < 4; ++i) {
        const int c = (ntb + i) * 16 + m;                 // col in [0,512)
        const int h = c >> 6, d = c & 63;
        for (int r = 0; r < 8; ++r) {
            int g = mt * 16 + half * 8 + r;               // row in [0, B*K)
            size_t bh = (size_t)(g >> 8) * HEADS + h;
            if (blockIdx.z == 0)
                Kz[bh * (DHEAD * KCTX) + bswz_off(d, g & 255, KCTX)] = (__bf16)acc[i][r];
            else
                Vz[bh * (KCTX * DHEAD) + bswz_off(g & 255, d, DHEAD)] = (__bf16)acc[i][r];
        }
    }
}

// ---------------------------------------------------------------- gathered Q projection
// qg[m,:] = xF[perm[m],:] @ Wq.  Block: 128 threads; wave computes 16x64.
__global__ void q_proj(const float* __restrict__ xF, const int* __restrict__ perm,
                       const __bf16* __restrict__ Wqz, __bf16* __restrict__ qg) {
    __shared__ __bf16 sA[2][16][40];
    const int tid = threadIdx.x;
    const int lane = tid & 31, wv = tid >> 5;
    const int m = lane & 15, half = lane >> 4;
    const int mt = blockIdx.y;                           // 0..4095
    const int ntb = blockIdx.x * 16 + wv * 4;            // 0,4,..,28

    const int srow = tid >> 3, scol = (tid & 7) * 4;
    const int prow = perm[mt * 16 + srow];
    const float* grow = xF + (size_t)prow * CH + scol;

    v8f acc[4] = {};
    {
        v4f x = *(const v4f*)(grow);
        v4bf y; y.x = (__bf16)x.x; y.y = (__bf16)x.y; y.z = (__bf16)x.z; y.w = (__bf16)x.w;
        *(v4bf*)&sA[0][srow][scol] = y;
    }
    const int NK = CH / 32;
    for (int kc = 0; kc < NK; ++kc) {
        __syncthreads();
        if (kc + 1 < NK) {
            v4f x = *(const v4f*)(grow + (kc + 1) * 32);
            v4bf y; y.x = (__bf16)x.x; y.y = (__bf16)x.y; y.z = (__bf16)x.z; y.w = (__bf16)x.w;
            *(v4bf*)&sA[(kc + 1) & 1][srow][scol] = y;
        }
        v16bf a = load_a_frag(&sA[kc & 1][m][0], half);
        for (int i = 0; i < 4; ++i) {
            v16bf b = load_b_frag(Wqz + (size_t)(kc * 32 + ntb + i) * 512, lane);
            acc[i] = __builtin_amdgcn_wmma_f32_16x16x32_bf16(false, a, false, b, (short)0, acc[i], false, false);
        }
    }
    for (int i = 0; i < 4; ++i)
        for (int r = 0; r < 8; ++r)
            qg[(size_t)(mt * 16 + half * 8 + r) * HD + (ntb + i) * 16 + m] = (__bf16)acc[i][r];
}

// ---------------------------------------------------------------- attention core
// One wave per (b, h, 16-query tile): S = scale*Q@K^T, softmax, O = P@V.
__global__ void attn_core(const __bf16* __restrict__ qg, const __bf16* __restrict__ Kz,
                          const __bf16* __restrict__ Vz, __bf16* __restrict__ o) {
    __shared__ float  sS[16][260];
    __shared__ __bf16 sP[16][264];
    const int lane = threadIdx.x & 31;
    const int m = lane & 15, half = lane >> 4;
    const int qt = blockIdx.x, h = blockIdx.y, b = blockIdx.z;
    const size_t qbase = (size_t)b * NPB + qt * 16;
    const size_t bh = (size_t)b * HEADS + h;
    const __bf16* Kzh = Kz + bh * (DHEAD * KCTX);        // swizzled, ktot=64,  ntot=256
    const __bf16* Vzh = Vz + bh * (KCTX * DHEAD);        // swizzled, ktot=256, ntot=64

    v16bf aq[2];
    for (int dc = 0; dc < 2; ++dc)
        aq[dc] = load_a_frag(qg + (qbase + m) * HD + h * DHEAD + dc * 32, half);

    // S = Q @ K^T
    for (int kt = 0; kt < KCTX / 16; ++kt) {
        v8f acc = {};
        for (int dc = 0; dc < 2; ++dc) {
            v16bf bk = load_b_frag(Kzh + (size_t)(dc * 16 + kt) * 512, lane);
            acc = __builtin_amdgcn_wmma_f32_16x16x32_bf16(false, aq[dc], false, bk, (short)0, acc, false, false);
        }
        for (int r = 0; r < 8; ++r)
            sS[half * 8 + r][kt * 16 + m] = 0.125f * acc[r];   // scale = D^-0.5
    }
    __syncthreads();

    // fp32 softmax over K=256, one row per lane 0..15
    if (lane < 16) {
        float mx = -3.0e38f;
        for (int k = 0; k < KCTX; ++k) mx = fmaxf(mx, sS[lane][k]);
        float sum = 0.f;
        for (int k = 0; k < KCTX; ++k) { float e = __expf(sS[lane][k] - mx); sS[lane][k] = e; sum += e; }
        float inv = 1.f / sum;
        for (int k = 0; k < KCTX; ++k) sP[lane][k] = (__bf16)(sS[lane][k] * inv);
    }
    __syncthreads();

    // O = P @ V : 4 accumulators cover all of D; A fragment shared per k-chunk
    v8f accs[4] = {};
    for (int kc = 0; kc < KCTX / 32; ++kc) {
        v16bf a = load_a_frag(&sP[m][kc * 32], half);
        for (int i = 0; i < 4; ++i) {
            v16bf bv = load_b_frag(Vzh + (size_t)(kc * 4 + i) * 512, lane);
            accs[i] = __builtin_amdgcn_wmma_f32_16x16x32_bf16(false, a, false, bv, (short)0, accs[i], false, false);
        }
    }
    for (int i = 0; i < 4; ++i)
        for (int r = 0; r < 8; ++r)
            o[(qbase + half * 8 + r) * HD + h * DHEAD + i * 16 + m] = (__bf16)accs[i][r];
}

// ---------------------------------------------------------------- output projection + scatter + residual
// out[perm[m], :] = o[m, :] @ Wout + b_out + xF[perm[m], :].  Wave computes 16x64.
__global__ void out_proj(const __bf16* __restrict__ o, const __bf16* __restrict__ Woz,
                         const float* __restrict__ b_out, const int* __restrict__ perm,
                         const float* __restrict__ xF, float* __restrict__ out) {
    const int lane = threadIdx.x & 31;
    const int m = lane & 15, half = lane >> 4;
    const int ntb = blockIdx.x * 4;                      // 0,4,8,12
    const int mt = blockIdx.y;                           // 0..4095

    int prow[8];
    for (int r = 0; r < 8; ++r) prow[r] = perm[mt * 16 + half * 8 + r];

    v8f acc[4] = {};
    for (int kc = 0; kc < HD / 32; ++kc) {
        v16bf a = load_a_frag(o + (size_t)(mt * 16 + m) * HD + kc * 32, half);
        for (int i = 0; i < 4; ++i) {
            v16bf b = load_b_frag(Woz + (size_t)(kc * 16 + ntb + i) * 512, lane);
            acc[i] = __builtin_amdgcn_wmma_f32_16x16x32_bf16(false, a, false, b, (short)0, acc[i], false, false);
        }
    }
    for (int i = 0; i < 4; ++i) {
        const int n = (ntb + i) * 16 + m;
        const float bo = b_out[n];
        for (int r = 0; r < 8; ++r) {
            const size_t p = (size_t)prow[r];
            out[p * CH + n] = acc[i][r] + bo + xF[p * CH + n];
        }
    }
}

// ---------------------------------------------------------------- launch
extern "C" void kernel_launch(void* const* d_in, const int* in_sizes, int n_in,
                              void* d_out, int out_size, void* d_ws, size_t ws_size,
                              hipStream_t stream) {
    const float* xF   = (const float*)d_in[0];
    const float* ctx  = (const float*)d_in[1];
    const int*   perm = (const int*)d_in[2];
    const float* Wq   = (const float*)d_in[3];
    const float* Wk   = (const float*)d_in[4];
    const float* Wv   = (const float*)d_in[5];
    const float* Wout = (const float*)d_in[6];
    const float* bo   = (const float*)d_in[7];
    float* out = (float*)d_out;

    char* ws = (char*)d_ws;
    size_t off = 0;
    __bf16* Wqz = (__bf16*)(ws + off); off += (size_t)CH * HD * 2;
    __bf16* Wkz = (__bf16*)(ws + off); off += (size_t)CTXD * HD * 2;
    __bf16* Wvz = (__bf16*)(ws + off); off += (size_t)CTXD * HD * 2;
    __bf16* Woz = (__bf16*)(ws + off); off += (size_t)HD * CH * 2;
    __bf16* Kz  = (__bf16*)(ws + off); off += (size_t)BATCH * HEADS * KCTX * DHEAD * 2;
    __bf16* Vz  = (__bf16*)(ws + off); off += (size_t)BATCH * HEADS * KCTX * DHEAD * 2;
    __bf16* qg  = (__bf16*)(ws + off); off += (size_t)N_PTS * HD * 2;
    __bf16* o_w = (__bf16*)(ws + off); off += (size_t)N_PTS * HD * 2;

    // 1) weights -> bf16, pre-swizzled into B column panels
    cast_swz_b<<<(CH * HD + 255) / 256, 256, 0, stream>>>(Wq, Wqz, CH, HD);
    cast_swz_b<<<(CTXD * HD + 255) / 256, 256, 0, stream>>>(Wk, Wkz, CTXD, HD);
    cast_swz_b<<<(CTXD * HD + 255) / 256, 256, 0, stream>>>(Wv, Wvz, CTXD, HD);
    cast_swz_b<<<(HD * CH + 255) / 256, 256, 0, stream>>>(Wout, Woz, HD, CH);

    // 2) K/V projections (z selects K vs V)
    kv_proj<<<dim3(HD / 256, (BATCH * KCTX) / 16, 2), 128, 0, stream>>>(ctx, Wkz, Wvz, Kz, Vz);

    // 3) gathered Q projection
    q_proj<<<dim3(HD / 256, N_PTS / 16), 128, 0, stream>>>(xF, perm, Wqz, qg);

    // 4) attention
    attn_core<<<dim3(NPB / 16, HEADS, BATCH), 32, 0, stream>>>(qg, Kz, Vz, o_w);

    // 5) output projection + bias + residual + inverse-perm scatter
    out_proj<<<dim3(CH / 64, N_PTS / 16), 32, 0, stream>>>(o_w, Woz, bo, perm, xF, out);
}
